// IterativeGraph_2534030704687
// MI455X (gfx1250) — matrix operand order repaired and checked by reading, not compiled, not run on hardware
//
#include <hip/hip_runtime.h>

typedef __attribute__((ext_vector_type(16))) _Float16 v16h;
typedef __attribute__((ext_vector_type(8)))  _Float16 v8h;
typedef __attribute__((ext_vector_type(8)))  float    v8f;

#define KJ    17            // joints
#define IN_F  19            // input features = 2 + 17
#define HID   38            // hidden width
#define EPSL  1e-5f
#define WAVES 2             // waves per block
#define SPW   16            // samples per wave (WMMA M dim)
#define SPB   (WAVES*SPW)   // samples per block = 32
#define NFRAG 13            // 3 (L1) + 6 (L2) + 4 (out) B-fragments

#define PLOC_SZ  (SPW*KJ*2)            // 544 floats
#define PD_OFF   PLOC_SZ               // pdist block starts here
#define PD_SZ    (SPW*KJ*KJ)           // 4624 floats
#define ZERO_OFF (PD_OFF + PD_SZ)      // 5168: shared zero slot
#define STATE_SZ 5184                  // padded

__device__ __forceinline__ v8f wmma16(v16h a, v16h b, v8f c) {
  // D = A(16x32 f16) * B(32x16 f16) + C(16x16 f32)
  return __builtin_amdgcn_wmma_f32_16x16x32_f16(
      false, a, false, b, (short)0, c, false, false);
}

// halves 0..7 <- lo (K = hi*8+i), halves 8..15 <- hf (K = 16+hi*8+i)
__device__ __forceinline__ v16h combine8(v8h lo, v8h hf) {
  return __builtin_shufflevector(lo, hf, 0,1,2,3,4,5,6,7,8,9,10,11,12,13,14,15);
}

__global__ __launch_bounds__(WAVES * 32)
void posegraph_wmma_kernel(
    const float* __restrict__ g_ploc, const float* __restrict__ g_pdist,
    const float* __restrict__ g_w1,  const float* __restrict__ g_ln1g, const float* __restrict__ g_ln1b,
    const float* __restrict__ g_w2,  const float* __restrict__ g_ln2g, const float* __restrict__ g_ln2b,
    const float* __restrict__ g_wc,  const float* __restrict__ g_wr,
    float* __restrict__ g_out)
{
  // B fragments pre-swizzled: lane l holds column n=l&15; halves i<8 hold K=(l>>4)*8+i, i>=8 hold K=16+(l>>4)*8+(i-8)
  __shared__ __attribute__((aligned(32))) _Float16 s_wfrag[NFRAG][32][16];
  __shared__ float s_ln[4][40];                    // ln1_g, ln1_b, ln2_g, ln2_b
  __shared__ __attribute__((aligned(16))) float s_state[WAVES][STATE_SZ]; // ploc | pdist | zero
  __shared__ __attribute__((aligned(16))) float s_h[WAVES][16][48];       // activations / logits
  __shared__ __attribute__((aligned(32))) _Float16 s_xa[WAVES][16][32];   // layer-1 A staging (f16)
  __shared__ __attribute__((aligned(32))) _Float16 s_xh[WAVES][16][64];   // layer-2/3 A staging (f16)
  __shared__ float s_stat[WAVES][2][16];
  __shared__ float s_mu[WAVES][16], s_rstd[WAVES][16];

  const int tid  = threadIdx.x;
  const int lane = tid & 31;
  const int w    = tid >> 5;
  const int l15  = lane & 15;   // A: sample row M; B/D: column N
  const int hi   = lane >> 4;

  // ---- stage weight fragments (block-cooperative) ----
  for (int idx = tid; idx < NFRAG * 512; idx += blockDim.x) {
    int f = idx >> 9, rem = idx & 511;
    int l = rem >> 4, i = rem & 15;
    int kk = (i < 8) ? ((l >> 4) * 8 + i) : (16 + (l >> 4) * 8 + (i - 8));   // K within 32-wide chunk
    int nl = l & 15;
    float v = 0.f;
    if (f < 3) {                                   // layer1: B[k][n] = W1[n][k]
      int n = f * 16 + nl;
      if (kk < IN_F && n < HID) v = g_w1[n * IN_F + kk];
    } else if (f < 9) {                            // layer2: 2 k-chunks x 3 n-tiles
      int c = (f - 3) / 3, t = (f - 3) % 3;
      int k = c * 32 + kk, n = t * 16 + nl;
      if (k < HID && n < HID) v = g_w2[n * HID + k];
    } else {                                       // out: [Wc;Wr], 2 k-chunks x 2 n-tiles
      int c = (f - 9) >> 1, t = (f - 9) & 1;
      int k = c * 32 + kk, n = t * 16 + nl;
      if (k < HID) {
        if (n < KJ)          v = g_wc[n * HID + k];
        else if (n < KJ + 2) v = g_wr[(n - KJ) * HID + k];
      }
    }
    s_wfrag[f][l][i] = (_Float16)v;
  }
  for (int idx = tid; idx < 4 * HID; idx += blockDim.x) {
    int r = idx / HID, c = idx % HID;
    const float* p = (r == 0) ? g_ln1g : (r == 1) ? g_ln1b : (r == 2) ? g_ln2g : g_ln2b;
    s_ln[r][c] = p[c];
  }

  // ---- stage per-wave mutable state ----
  const int base = blockIdx.x * SPB + w * SPW;
  for (int idx = lane; idx < PLOC_SZ; idx += 32)
    s_state[w][idx] = g_ploc[base * KJ * 2 + idx];
  for (int idx = lane; idx < PD_SZ; idx += 32)
    s_state[w][PD_OFF + idx] = g_pdist[base * KJ * KJ + idx];
  if (lane == 0) s_state[w][ZERO_OFF] = 0.f;

  __syncthreads();

  // ---- hoist all B fragments into registers (loop-invariant) ----
  v16h B[NFRAG];
#pragma unroll
  for (int f = 0; f < NFRAG; ++f)
    B[f] = *(const v16h*)&s_wfrag[f][lane][0];

  // D tiles -> s_h rows (lane: n=l15, vgpr r -> M = hi*8+r)
  auto store_tiles3 = [&](v8f c0, v8f c1, v8f c2) {
#pragma unroll
    for (int r = 0; r < 8; ++r) {
      int mm = hi * 8 + r;
      s_h[w][mm][l15]      = c0[r];
      s_h[w][mm][16 + l15] = c1[r];
      s_h[w][mm][32 + l15] = c2[r];
    }
  };

  // branch-free LN stats: lanes 0-15 sum x, lanes 16-31 sum x^2 (cols 38..47 are exact 0)
  auto ln_stats = [&]() {
    __builtin_amdgcn_wave_barrier();
    float acc = 0.f;
#pragma unroll
    for (int k4 = 0; k4 < 48; k4 += 4) {
      float4 x = *(const float4*)&s_h[w][l15][k4];
      acc = fmaf(x.x, hi ? x.x : 1.f, acc);
      acc = fmaf(x.y, hi ? x.y : 1.f, acc);
      acc = fmaf(x.z, hi ? x.z : 1.f, acc);
      acc = fmaf(x.w, hi ? x.w : 1.f, acc);
    }
    s_stat[w][hi][l15] = acc;
    __builtin_amdgcn_wave_barrier();
    if (lane < 16) {
      float mu = s_stat[w][0][lane] * (1.f / (float)HID);
      float q  = s_stat[w][1][lane] * (1.f / (float)HID);
      s_mu[w][lane]   = mu;
      s_rstd[w][lane] = rsqrtf(q - mu * mu + EPSL);
    }
    __builtin_amdgcn_wave_barrier();
  };

  // relu(LN(H)) -> s_xh (f16), each lane handles cols {lane, lane+32}
  auto ln_relu_stage = [&](int gi) {
    int  k2  = lane + 32;
    bool k2v = (k2 < HID);
    float gA = s_ln[gi][lane],             bA = s_ln[gi + 1][lane];
    float gB = k2v ? s_ln[gi][k2] : 0.f,   bB = k2v ? s_ln[gi + 1][k2] : 0.f;
#pragma unroll
    for (int m2 = 0; m2 < 16; ++m2) {
      float mu = s_mu[w][m2], rs = s_rstd[w][m2];
      float x0 = s_h[w][m2][lane];
      float v0 = fmaxf((x0 - mu) * rs * gA + bA, 0.f);
      float x1 = k2v ? s_h[w][m2][k2] : 0.f;
      float v1 = k2v ? fmaxf((x1 - mu) * rs * gB + bB, 0.f) : 0.f;
      s_xh[w][m2][lane]      = (_Float16)v0;
      s_xh[w][m2][lane + 32] = (_Float16)v1;
    }
    __builtin_amdgcn_wave_barrier();
  };

  // A fragment from staged f16 buffer: two contiguous ds_load_b128 per chunk
  auto afrag_xh = [&](int c) {
    return combine8(*(const v8h*)&s_xh[w][l15][c * 32 + hi * 8],
                    *(const v8h*)&s_xh[w][l15][c * 32 + 16 + hi * 8]);
  };

  // ---- 17 sequential decode steps ----
#pragma unroll 1
  for (int kid = 0; kid < KJ; ++kid) {
    // stage x (f16, padded to 32) branch-free: per-lane offset+stride resolved once
    {
      int k = lane, off, stride;
      if (k < 2)        { off = kid * 2 + k;                      stride = KJ * 2;  }
      else if (k < IN_F){ off = PD_OFF + kid * KJ + (k - 2);      stride = KJ * KJ; }
      else              { off = ZERO_OFF;                         stride = 0;       }
#pragma unroll
      for (int m2 = 0; m2 < 16; ++m2) {
        s_xa[w][m2][lane] = (_Float16)s_state[w][off];
        off += stride;
      }
    }
    __builtin_amdgcn_wave_barrier();

    // ---- layer 1: H1 = X @ W1^T ----
    v16h a1 = combine8(*(const v8h*)&s_xa[w][l15][hi * 8],
                       *(const v8h*)&s_xa[w][l15][16 + hi * 8]);
    v8f h0 = {}, h1 = {}, h2 = {};
    h0 = wmma16(a1, B[0], h0);
    h1 = wmma16(a1, B[1], h1);
    h2 = wmma16(a1, B[2], h2);
    store_tiles3(h0, h1, h2);
    ln_stats();
    ln_relu_stage(0);

    // ---- layer 2: H2 = h @ W2^T ----
    v16h a20 = afrag_xh(0), a21 = afrag_xh(1);
    v8f g0 = {}, g1 = {}, g2 = {};
    g0 = wmma16(a20, B[3], g0);
    g1 = wmma16(a20, B[4], g1);
    g2 = wmma16(a20, B[5], g2);
    g0 = wmma16(a21, B[6], g0);
    g1 = wmma16(a21, B[7], g1);
    g2 = wmma16(a21, B[8], g2);
    store_tiles3(g0, g1, g2);
    ln_stats();
    ln_relu_stage(2);

    // ---- output head: cols 0..16 logits, 17..18 regression ----
    v16h a30 = afrag_xh(0), a31 = afrag_xh(1);
    v8f o0 = {}, o1 = {};
    o0 = wmma16(a30, B[9],  o0);
    o1 = wmma16(a30, B[10], o1);
    o0 = wmma16(a31, B[11], o0);
    o1 = wmma16(a31, B[12], o1);
#pragma unroll
    for (int r = 0; r < 8; ++r) {
      int mm = hi * 8 + r;
      s_h[w][mm][l15]      = o0[r];
      s_h[w][mm][16 + l15] = o1[r];
    }
    __builtin_amdgcn_wave_barrier();

    // ---- state update: sample l15, both half-waves cooperate ----
    {
      float best = s_h[w][l15][0];
      int   bi   = 0;
#pragma unroll
      for (int j = 1; j < KJ; ++j) {          // first-max semantics (jnp.argmax)
        float v = s_h[w][l15][j];
        if (v > best) { best = v; bi = j; }
      }
      float rx = s_h[w][l15][17], ry = s_h[w][l15][18];
      float* pl = &s_state[w][l15 * KJ * 2];
      float* pd = &s_state[w][PD_OFF + l15 * KJ * KJ];
      if (!hi) { pl[bi * 2 + 0] = rx; pl[bi * 2 + 1] = ry; }
      __builtin_amdgcn_wave_barrier();
#pragma unroll
      for (int jj = 0; jj < 9; ++jj) {        // hi=0: j 0..8, hi=1: j 9..16
        int j = jj + hi * 9;
        if (j < KJ) {
          float dx = pl[j * 2 + 0] - rx, dy = pl[j * 2 + 1] - ry;
          float sq = dx * dx + dy * dy;
          float d  = (sq > 0.f) ? sqrtf(sq) : 0.f;
          pd[j * KJ + bi] = d;
          pd[bi * KJ + j] = d;
        }
      }
    }
    __builtin_amdgcn_wave_barrier();
  }

  // ---- write final ploc ----
  for (int idx = lane; idx < PLOC_SZ; idx += 32)
    g_out[base * KJ * 2 + idx] = s_state[w][idx];
}

extern "C" void kernel_launch(void* const* d_in, const int* in_sizes, int n_in,
                              void* d_out, int out_size, void* d_ws, size_t ws_size,
                              hipStream_t stream) {
  const float* ploc  = (const float*)d_in[0];
  const float* pdist = (const float*)d_in[1];
  // d_in[2] = gt_adj (unused; adjacency never affects the returned ploc)
  const float* W1   = (const float*)d_in[3];
  const float* ln1g = (const float*)d_in[4];
  const float* ln1b = (const float*)d_in[5];
  const float* W2   = (const float*)d_in[6];
  const float* ln2g = (const float*)d_in[7];
  const float* ln2b = (const float*)d_in[8];
  const float* Wc   = (const float*)d_in[9];
  const float* Wr   = (const float*)d_in[10];
  float* out = (float*)d_out;

  const int n = in_sizes[0] / (KJ * 2);   // 32768 samples
  const int blocks = n / SPB;             // 32 samples per block (2 waves x 16)
  posegraph_wmma_kernel<<<blocks, WAVES * 32, 0, stream>>>(
      ploc, pdist, W1, ln1g, ln1b, W2, ln2g, ln2b, Wc, Wr, out);
}
